// BWGNN_7138235646046
// MI455X (gfx1250) — matrix operand order repaired and checked by reading, not compiled, not run on hardware
//
#include <hip/hip_runtime.h>
#include <hip/hip_bf16.h>

// ---------------------------------------------------------------------------
// BWGNN forward for MI455X (gfx1250, wave32, WMMA).
// - Dense layers: v_wmma_f32_16x16x32_bf16, weights staged in LDS via
//   global_load_async_to_lds_b128 (ASYNCcnt) and read back with ds_load_b128.
// - SpMM: edge-parallel float4 L2 atomics (h matrix is L2-resident: 25.6 MB
//   vs 192 MB L2).
// - Theta/concat layer fused into the W3 GEMM A-operand (never materialize
//   the N x 192 matrix).
// ---------------------------------------------------------------------------

typedef __attribute__((ext_vector_type(16))) __bf16 v16bf;
typedef __attribute__((ext_vector_type(8)))  unsigned v8u;   // 16 bf16 as dwords
typedef __attribute__((ext_vector_type(8)))  float v8f;

#define LD4(p) (*(const float4*)(p))

// fp32 -> bf16 pair packed into one dword (low = first).
// Hardware v_cvt_pk_bf16_f32 when the toolchain exposes it; otherwise
// round-half-up via pure dword integer ops (2 VALU/elem, no 16-bit shuffles).
static __device__ __forceinline__ unsigned cvt2bf(float x, float y) {
#if __has_builtin(__builtin_amdgcn_cvt_pk_bf16_f32)
  auto r = __builtin_amdgcn_cvt_pk_bf16_f32(x, y);   // v_cvt_pk_bf16_f32
  unsigned u;
  __builtin_memcpy(&u, &r, 4);
  return u;
#else
  unsigned ux = __float_as_uint(x) + 0x8000u;
  unsigned uy = __float_as_uint(y) + 0x8000u;
  return (ux >> 16) | (uy & 0xFFFF0000u);           // lshr + v_and_or_b32
#endif
}

static __device__ __forceinline__ void packf4(v8u& u, int base, float4 a, float4 b) {
  u[base + 0] = cvt2bf(a.x, a.y);
  u[base + 1] = cvt2bf(a.z, a.w);
  u[base + 2] = cvt2bf(b.x, b.y);
  u[base + 3] = cvt2bf(b.z, b.w);
}

static __device__ __forceinline__ float4 f4_axpby(float a, const float4& x,
                                                  float b, const float4& y) {
  return make_float4(a * x.x + b * y.x, a * x.y + b * y.y,
                     a * x.z + b * y.z, a * x.w + b * y.w);
}
static __device__ __forceinline__ float4 f4_scale(float a, const float4& x) {
  return make_float4(a * x.x, a * x.y, a * x.z, a * x.w);
}

// Async global -> LDS stage of `nfloats` (multiple of 4*blockDim) from src.
// Every lane moves 16B per iteration on the ASYNCcnt path; then waits ASYNCcnt 0 and barriers itself.
static __device__ __forceinline__ void stage_to_lds(float* lds, const float* src,
                                                    int nfloats) {
  for (int idx = threadIdx.x * 4; idx < nfloats; idx += blockDim.x * 4) {
    unsigned dst = (unsigned)(unsigned long long)(&lds[idx]);  // LDS byte offset
    const float* g = src + idx;
    asm volatile("global_load_async_to_lds_b128 %0, %1, off"
                 :: "v"(dst), "v"(g) : "memory");
  }
#if __has_builtin(__builtin_amdgcn_s_wait_asynccnt)
  __builtin_amdgcn_s_wait_asynccnt(0);
#else
  asm volatile("s_wait_asynccnt 0" ::: "memory");
#endif
  __syncthreads();
}

// --------------------------- utility kernels -------------------------------

__global__ void zero_f32(float* __restrict__ p, long long n) {
  long long i = blockIdx.x * (long long)blockDim.x + threadIdx.x;
  long long stride = (long long)gridDim.x * blockDim.x;
  for (; i < n; i += stride) p[i] = 0.0f;
}

__global__ void degree_accum(const long long* __restrict__ rows,
                             float* __restrict__ deg, long long E) {
  long long i = blockIdx.x * (long long)blockDim.x + threadIdx.x;
  if (i < E) atomicAdd(&deg[rows[i]], 1.0f);
}

__global__ void deg_to_dinv(float* __restrict__ d, int n) {
  int i = blockIdx.x * blockDim.x + threadIdx.x;
  if (i < n) {
    float v = d[i];
    d[i] = (v > 0.0f) ? rsqrtf(v) : 0.0f;
  }
}

// acc[row] += dinv[row]*dinv[col] * feat[col]  (64 features, float4 per thread)
__global__ void spmm_edges(const long long* __restrict__ rows,
                           const long long* __restrict__ cols,
                           const float* __restrict__ dinv,
                           const float* __restrict__ feat,
                           float* __restrict__ acc, long long E) {
  long long t = blockIdx.x * (long long)blockDim.x + threadIdx.x;
  long long e = t >> 4;
  if (e >= E) return;
  int c4 = (int)(t & 15) * 4;
  long long r = rows[e], c = cols[e];
  float w = dinv[r] * dinv[c];
  float4 v = LD4(feat + c * 64 + c4);
  float* dst = acc + r * 64 + c4;
  atomicAdd(dst + 0, w * v.x);
  atomicAdd(dst + 1, w * v.y);
  atomicAdd(dst + 2, w * v.z);
  atomicAdd(dst + 3, w * v.w);
}

__global__ void sub_f32(const float* __restrict__ a, const float* __restrict__ b,
                        float* __restrict__ o, long long n) {
  long long i = blockIdx.x * (long long)blockDim.x + threadIdx.x;
  if (i < n) o[i] = a[i] - b[i];
}

// ------------------------ dense WMMA GEMM layers ---------------------------
// Y[N x O] = act(X[N x K] @ W[O x K]^T + bias), bf16 inputs into f32 WMMA.
// One wave computes a 16x16 tile; block = 8 waves = 2 row tiles x (O/16) col.
// Weight matrix staged once per block into LDS via async loads.
template <int K, int O, bool RELU>
__global__ __launch_bounds__(256) void gemm_bf16_wmma(
    const float* __restrict__ X, const float* __restrict__ W,
    const float* __restrict__ bias, float* __restrict__ Y, int Nrows) {
  constexpr int CT = O / 16;   // col tiles covered by the block
  constexpr int RT = 8 / CT;   // row tiles covered by the block
  __shared__ float wlds[O * K];
  stage_to_lds(wlds, W, O * K);

  const int wave = threadIdx.x >> 5;
  const int lane = threadIdx.x & 31;
  const int wr = wave / CT, wc = wave % CT;
  const int row0 = blockIdx.x * (RT * 16) + wr * 16;
  if (row0 >= Nrows) return;
  const int col0 = wc * 16;

  const int  m   = lane & 15;
  const bool lo  = lane < 16;
  const float* xr = X + (size_t)(row0 + m) * K;       // A row for this lane
  const float* wn = wlds + (size_t)(col0 + m) * K;    // B column n = m (LDS)
  const int kAoff = lo ? 0 : 8;                       // A K-half select
  const int kBoff = lo ? 0 : 8;                       // B K-half select

  v8f c = {};
#pragma unroll
  for (int k0 = 0; k0 < K; k0 += 32) {
    v8u au, bu;
    {   // A 16x32 tile: halves 0-7 = K kA..kA+7, halves 8-15 = K kA+16..kA+23
      const float* pa = xr + k0 + kAoff;
      packf4(au, 0, LD4(pa), LD4(pa + 4));
      packf4(au, 4, LD4(pa + 16), LD4(pa + 20));
    }
    {   // B 32x16 tile: halves 0-7 = K kB..kB+7, halves 8-15 = K kB+16..kB+23 (ds_load_b128)
      const float* pb = wn + k0 + kBoff;
      packf4(bu, 0, LD4(pb), LD4(pb + 4));
      packf4(bu, 4, LD4(pb + 16), LD4(pb + 20));
    }
    v16bf a = __builtin_bit_cast(v16bf, au);
    v16bf b = __builtin_bit_cast(v16bf, bu);
    c = __builtin_amdgcn_wmma_f32_16x16x32_bf16(false, a, false, b,
                                                (short)0, c, false, false);
  }

  const int rbase = row0 + (lo ? 0 : 8);
  const int n = col0 + m;
  const float bn = bias[n];
#pragma unroll
  for (int j = 0; j < 8; ++j) {
    float v = c[j] + bn;
    if (RELU) v = v > 0.0f ? v : 0.0f;
    Y[(size_t)(rbase + j) * O + n] = v;
  }
}

// Fused Bernstein-theta GEMM: A(N x 192) is built on the fly from h,p1,p2
//   cols [0,64):   3*h - 3*p1 + 0.75*p2
//   cols [64,128): 3*p1 - 1.5*p2
//   cols [128,192):0.75*p2
// Y = relu(A @ W3[64 x 192]^T + b3)
__global__ __launch_bounds__(256) void poly_gemm_wmma(
    const float* __restrict__ H, const float* __restrict__ P1,
    const float* __restrict__ P2, const float* __restrict__ W,
    const float* __restrict__ bias, float* __restrict__ Y, int Nrows) {
  constexpr int K = 192, O = 64;
  constexpr int CT = O / 16;
  constexpr int RT = 8 / CT;
  __shared__ float wlds[O * K];
  stage_to_lds(wlds, W, O * K);

  const int wave = threadIdx.x >> 5;
  const int lane = threadIdx.x & 31;
  const int wr = wave / CT, wc = wave % CT;
  const int row0 = blockIdx.x * (RT * 16) + wr * 16;
  if (row0 >= Nrows) return;
  const int col0 = wc * 16;

  const int  m  = lane & 15;
  const bool lo = lane < 16;
  const float* hrow  = H  + (size_t)(row0 + m) * 64;
  const float* p1row = P1 + (size_t)(row0 + m) * 64;
  const float* p2row = P2 + (size_t)(row0 + m) * 64;
  const float* wn = wlds + (size_t)(col0 + m) * K;
  const int kAoff = lo ? 0 : 8;
  const int kBoff = lo ? 0 : 8;

  v8f c = {};
#pragma unroll
  for (int k0 = 0; k0 < K; k0 += 32) {
    const int blk = k0 >> 6;           // constant-folded per unrolled iter
    const int kk0 = (k0 & 63) + kAoff; // offset inside the 64-wide block
    v8u au, bu;
#pragma unroll
    for (int g = 0; g < 2; ++g) {      // halves 0-7 then 8-15 (K +16)
      const int kb = kk0 + g * 16;
      float4 va, vb;
      if (blk == 0) {
        float4 x0 = LD4(hrow + kb),  x1 = LD4(hrow + kb + 4);
        float4 y0 = LD4(p1row + kb), y1 = LD4(p1row + kb + 4);
        float4 z0 = LD4(p2row + kb), z1 = LD4(p2row + kb + 4);
        va = f4_axpby(1.0f, f4_axpby(3.0f, x0, -3.0f, y0), 0.75f, z0);
        vb = f4_axpby(1.0f, f4_axpby(3.0f, x1, -3.0f, y1), 0.75f, z1);
      } else if (blk == 1) {
        float4 y0 = LD4(p1row + kb), y1 = LD4(p1row + kb + 4);
        float4 z0 = LD4(p2row + kb), z1 = LD4(p2row + kb + 4);
        va = f4_axpby(3.0f, y0, -1.5f, z0);
        vb = f4_axpby(3.0f, y1, -1.5f, z1);
      } else {
        va = f4_scale(0.75f, LD4(p2row + kb));
        vb = f4_scale(0.75f, LD4(p2row + kb + 4));
      }
      packf4(au, g * 4, va, vb);
    }
    {
      const float* pb = wn + k0 + kBoff;
      packf4(bu, 0, LD4(pb), LD4(pb + 4));
      packf4(bu, 4, LD4(pb + 16), LD4(pb + 20));
    }
    v16bf a = __builtin_bit_cast(v16bf, au);
    v16bf b = __builtin_bit_cast(v16bf, bu);
    c = __builtin_amdgcn_wmma_f32_16x16x32_bf16(false, a, false, b,
                                                (short)0, c, false, false);
  }

  const int rbase = row0 + (lo ? 0 : 8);
  const int n = col0 + m;
  const float bn = bias[n];
#pragma unroll
  for (int j = 0; j < 8; ++j) {
    float v = c[j] + bn;
    v = v > 0.0f ? v : 0.0f;
    Y[(size_t)(rbase + j) * O + n] = v;
  }
}

// out[N x 2] = h3[N x 64] @ W4[2 x 64]^T + b4   (tiny N-dim: plain VALU)
__global__ void final_linear(const float* __restrict__ h3,
                             const float* __restrict__ W4,
                             const float* __restrict__ b4,
                             float* __restrict__ out, int n) {
  int i = blockIdx.x * blockDim.x + threadIdx.x;
  if (i >= n) return;
  const float* hr = h3 + (size_t)i * 64;
  float a0 = b4[0], a1 = b4[1];
#pragma unroll
  for (int k = 0; k < 64; k += 4) {
    float4 v  = LD4(hr + k);
    float4 w0 = LD4(W4 + k);
    float4 w1 = LD4(W4 + 64 + k);
    a0 += v.x * w0.x + v.y * w0.y + v.z * w0.z + v.w * w0.w;
    a1 += v.x * w1.x + v.y * w1.y + v.z * w1.z + v.w * w1.w;
  }
  out[2 * i + 0] = a0;
  out[2 * i + 1] = a1;
}

// ------------------------------- driver ------------------------------------

extern "C" void kernel_launch(void* const* d_in, const int* in_sizes, int n_in,
                              void* d_out, int out_size, void* d_ws, size_t ws_size,
                              hipStream_t stream) {
  const float*     in_feat = (const float*)d_in[0];
  const long long* ei      = (const long long*)d_in[1];   // int64 edge_index [2,E]
  const float* W1 = (const float*)d_in[2];
  const float* b1 = (const float*)d_in[3];
  const float* W2 = (const float*)d_in[4];
  const float* b2 = (const float*)d_in[5];
  const float* W3 = (const float*)d_in[6];
  const float* b3 = (const float*)d_in[7];
  const float* W4 = (const float*)d_in[8];
  const float* b4 = (const float*)d_in[9];
  float* out = (float*)d_out;

  const int       N = in_sizes[0] / 128;   // 100000 (divisible by 32)
  const long long E = in_sizes[1] / 2;     // 1600000
  const long long* rows = ei;
  const long long* cols = ei + E;

  // workspace layout (floats): dinv[N] | h[N*64] | acc[N*64] | p1[N*64] | p2[N*64]
  float* ws   = (float*)d_ws;
  float* dinv = ws;
  float* h    = dinv + N;
  float* acc  = h  + (size_t)N * 64;
  float* p1   = acc + (size_t)N * 64;
  float* p2   = p1 + (size_t)N * 64;
  const long long NF = (long long)N * 64;

  // 1) degree -> dinv = deg>0 ? 1/sqrt(deg) : 0
  zero_f32<<<1024, 256, 0, stream>>>(dinv, (long long)N);
  degree_accum<<<(unsigned)((E + 255) / 256), 256, 0, stream>>>(rows, dinv, E);
  deg_to_dinv<<<(N + 255) / 256, 256, 0, stream>>>(dinv, N);

  // 2) MLP front: h1 -> acc, then h = relu(h1 @ W2^T + b2)
  const int gG = (N + 31) / 32;   // 32 rows per block
  gemm_bf16_wmma<128, 64, true><<<gG, 256, 0, stream>>>(in_feat, W1, b1, acc, N);
  gemm_bf16_wmma<64, 64, true><<<gG, 256, 0, stream>>>(acc, W2, b2, h, N);

  // 3) p1 = h - A_norm h ; p2 = p1 - A_norm p1
  const unsigned gS = (unsigned)((E * 16 + 255) / 256);
  const unsigned gV = (unsigned)((NF + 255) / 256);
  zero_f32<<<1024, 256, 0, stream>>>(acc, NF);
  spmm_edges<<<gS, 256, 0, stream>>>(rows, cols, dinv, h, acc, E);
  sub_f32<<<gV, 256, 0, stream>>>(h, acc, p1, NF);
  zero_f32<<<1024, 256, 0, stream>>>(acc, NF);
  spmm_edges<<<gS, 256, 0, stream>>>(rows, cols, dinv, p1, acc, E);
  sub_f32<<<gV, 256, 0, stream>>>(p1, acc, p2, NF);

  // 4) fused theta-concat GEMM -> h3 (reuse acc), then 64->2 projection
  poly_gemm_wmma<<<gG, 256, 0, stream>>>(h, p1, p2, W3, b3, acc, N);
  final_linear<<<(N + 255) / 256, 256, 0, stream>>>(acc, W4, b4, out, N);
}
